// TransformerBlock_49297634623628
// MI455X (gfx1250) — matrix operand order repaired and checked by reading, not compiled, not run on hardware
//
#include <hip/hip_runtime.h>
#include <hip/hip_bf16.h>
#include <math.h>

// ---------------------------------------------------------------------------
// Types for CDNA5 WMMA (wave32): v_wmma_f32_16x16x32_f16
// ---------------------------------------------------------------------------
typedef _Float16 v8h  __attribute__((ext_vector_type(8)));
typedef _Float16 v16h __attribute__((ext_vector_type(16)));
typedef float    v8f  __attribute__((ext_vector_type(8)));

#define D_MODEL 1024
#define D_FF    4096
#define T_SEQ   2048
#define NB      2
#define NHEAD   16
#define DKH     64
#define MTOT    (NB * T_SEQ)   // 4096 rows

// ---------------------------------------------------------------------------
// WMMA wrapper
// ---------------------------------------------------------------------------
__device__ __forceinline__ v8f wmma_f16(v16h a, v16h b, v8f c) {
    return __builtin_amdgcn_wmma_f32_16x16x32_f16(
        /*neg_a=*/false, a, /*neg_b=*/false, b,
        /*c_mod=*/(short)0, c, /*reuse_a=*/false, /*reuse_b=*/false);
}

// A-matrix fragment (16x32 f16). Per ISA: lanes 0-15 hold K={0..7,16..23},
// lanes 16-31 hold K={8..15,24..31}. p = start of this row's 32 K values.
__device__ __forceinline__ v16h frag_a(const _Float16* p, int hv) {
    v8h lo = *(const v8h*)(p + hv * 8);
    v8h hi = *(const v8h*)(p + 16 + hv * 8);
    return __builtin_shufflevector(lo, hi, 0,1,2,3,4,5,6,7,8,9,10,11,12,13,14,15);
}

// B-matrix fragment (32x16 f16, column n per lane). Same K order as A:
// lanes 0-15 hold K={0..7,16..23}, lanes 16-31 hold K={8..15,24..31}.
// p = start of column-n's 32 K values (B stored transposed, K-contiguous).
__device__ __forceinline__ v16h frag_b(const _Float16* p, int hv) {
    v8h lo = *(const v8h*)(p + hv * 8);
    v8h hi = *(const v8h*)(p + 16 + hv * 8);
    return __builtin_shufflevector(lo, hi, 0,1,2,3,4,5,6,7,8,9,10,11,12,13,14,15);
}

// LDS byte offset of a generic pointer to __shared__ (ISA: LDS aperture
// addresses truncate to addr[31:0]).
__device__ __forceinline__ unsigned lds_off(const void* p) {
    return (unsigned)(unsigned long long)p;
}

// ---------------------------------------------------------------------------
// Weight transpose + fp32 -> f16 convert:  src[K][N] -> dst[N][K]
// ---------------------------------------------------------------------------
__global__ __launch_bounds__(256) void transpose_f32_to_f16(
    const float* __restrict__ src, _Float16* __restrict__ dst, int K, int N) {
    __shared__ float tile[32][33];
    int n0 = blockIdx.x * 32, k0 = blockIdx.y * 32;
    int tx = threadIdx.x, ty = threadIdx.y;           // block (32,8)
    for (int i = 0; i < 32; i += 8)
        tile[ty + i][tx] = src[(size_t)(k0 + ty + i) * N + n0 + tx];
    __syncthreads();
    for (int i = 0; i < 32; i += 8)
        dst[(size_t)(n0 + ty + i) * K + k0 + tx] = (_Float16)tile[tx][ty + i];
}

// ---------------------------------------------------------------------------
// LayerNorm (fp32 in) -> f16 out, one row (1024 cols) per block of 256
// ---------------------------------------------------------------------------
__global__ __launch_bounds__(256) void layernorm_f16(
    const float* __restrict__ x, const float* __restrict__ g,
    const float* __restrict__ be, _Float16* __restrict__ out) {
    int row = blockIdx.x;
    const float4* xr = (const float4*)(x + (size_t)row * D_MODEL);
    float4 v = xr[threadIdx.x];
    float s  = v.x + v.y + v.z + v.w;
    float s2 = v.x * v.x + v.y * v.y + v.z * v.z + v.w * v.w;
    __shared__ float red[512];
    red[threadIdx.x] = s;
    red[256 + threadIdx.x] = s2;
    __syncthreads();
    for (int off = 128; off > 0; off >>= 1) {
        if ((int)threadIdx.x < off) {
            red[threadIdx.x]       += red[threadIdx.x + off];
            red[256 + threadIdx.x] += red[256 + threadIdx.x + off];
        }
        __syncthreads();
    }
    float mu   = red[0] * (1.0f / D_MODEL);
    float var  = red[256] * (1.0f / D_MODEL) - mu * mu;
    float rstd = rsqrtf(var + 1e-5f);
    int c = threadIdx.x * 4;
    float4 gv = ((const float4*)g)[threadIdx.x];
    float4 bv = ((const float4*)be)[threadIdx.x];
    _Float16* o = out + (size_t)row * D_MODEL + c;
    o[0] = (_Float16)((v.x - mu) * rstd * gv.x + bv.x);
    o[1] = (_Float16)((v.y - mu) * rstd * gv.y + bv.y);
    o[2] = (_Float16)((v.z - mu) * rstd * gv.z + bv.z);
    o[3] = (_Float16)((v.w - mu) * rstd * gv.w + bv.w);
}

// ---------------------------------------------------------------------------
// Generic WMMA GEMM:  C[M][N] = A[M][K](f16) * BT[N][K](f16)^T + bias
// MODE 0: store f16      MODE 1: exact GELU -> f16     MODE 2: +resid -> f32
// Block tile 64x64, K-step 32, 8 waves. Double-buffered LDS staging via
// GLOBAL_LOAD_ASYNC_TO_LDS_B128 (ASYNCcnt) so tile t+1 streams while t computes.
// ---------------------------------------------------------------------------
#define LDST 40   // padded LDS row stride in halfs (80B: 16B-aligned, bank-spread)

template <int MODE>
__global__ __launch_bounds__(256) void gemm_wmma(
    const _Float16* __restrict__ A, const _Float16* __restrict__ BT,
    const float* __restrict__ bias, const float* __restrict__ resid,
    void* __restrict__ out, int M, int N, int K) {
    __shared__ __align__(16) _Float16 As[2][64 * LDST];
    __shared__ __align__(16) _Float16 Bs[2][64 * LDST];
    int n0 = blockIdx.x * 64, m0 = blockIdx.y * 64;
    int tid = threadIdx.x;
    int lane = tid & 31, wave = tid >> 5;
    int wm = wave & 3, wn2 = wave >> 2;
    int nl = lane & 15, hv = lane >> 4;
    int lr = tid >> 2, lc = (tid & 3) * 8;   // cooperative tile load: one b128/thread

    // Per-thread LDS destinations (byte offsets) and global row bases (bytes)
    unsigned ldsA0 = lds_off(&As[0][lr * LDST + lc]);
    unsigned ldsA1 = lds_off(&As[1][lr * LDST + lc]);
    unsigned ldsB0 = lds_off(&Bs[0][lr * LDST + lc]);
    unsigned ldsB1 = lds_off(&Bs[1][lr * LDST + lc]);
    unsigned gArow = (unsigned)(((size_t)(m0 + lr) * K + lc) * 2);
    unsigned gBrow = (unsigned)(((size_t)(n0 + lr) * K + lc) * 2);

    // Async-copy one 64x32 tile chunk (this thread's 16B) of A and B into buf.
    auto issue = [&](int buf, int k0) {
        unsigned ga = gArow + (unsigned)(k0 * 2);
        unsigned gb = gBrow + (unsigned)(k0 * 2);
        unsigned la = buf ? ldsA1 : ldsA0;
        unsigned lb = buf ? ldsB1 : ldsB0;
        asm volatile("global_load_async_to_lds_b128 %0, %1, %2"
                     :: "v"(la), "v"(ga), "s"(A) : "memory");
        asm volatile("global_load_async_to_lds_b128 %0, %1, %2"
                     :: "v"(lb), "v"(gb), "s"(BT) : "memory");
    };

    v8f acc0 = {}; v8f acc1 = {};
    issue(0, 0);
    int nt = K / 32;
    for (int t = 0; t < nt; ++t) {
        int cur = t & 1;
        if (t + 1 < nt) {
            issue(cur ^ 1, (t + 1) * 32);
            // 2 ops for tile t+1 may remain in flight; tile t is complete.
            asm volatile("s_wait_asynccnt 0x2" ::: "memory");
        } else {
            asm volatile("s_wait_asynccnt 0x0" ::: "memory");
        }
        __syncthreads();
        v16h af = frag_a(&As[cur][(wm * 16 + nl) * LDST], hv);
        v16h b0 = frag_b(&Bs[cur][((wn2 * 2 + 0) * 16 + nl) * LDST], hv);
        v16h b1 = frag_b(&Bs[cur][((wn2 * 2 + 1) * 16 + nl) * LDST], hv);
        acc0 = wmma_f16(af, b0, acc0);
        acc1 = wmma_f16(af, b1, acc1);
        __syncthreads();
    }
    for (int t = 0; t < 2; ++t) {
        v8f acc = t ? acc1 : acc0;
        int n = n0 + (wn2 * 2 + t) * 16 + nl;
        float bi = bias[n];
        for (int j = 0; j < 8; ++j) {
            int m = m0 + wm * 16 + j + 8 * hv;
            float v = acc[j] + bi;
            if (MODE == 0) {
                ((_Float16*)out)[(size_t)m * N + n] = (_Float16)v;
            } else if (MODE == 1) {
                float gl = 0.5f * v * (1.0f + erff(v * 0.70710678118654752f));
                ((_Float16*)out)[(size_t)m * N + n] = (_Float16)gl;
            } else {
                ((float*)out)[(size_t)m * N + n] = v + resid[(size_t)m * N + n];
            }
        }
    }
}

// ---------------------------------------------------------------------------
// Extract V^T per (b,h): vT[(bh*64+d)][t] = qkv[b*T+t][2*D + h*64 + d]  (f16)
// ---------------------------------------------------------------------------
__global__ __launch_bounds__(256) void extract_vT(
    const _Float16* __restrict__ qkv, _Float16* __restrict__ vT) {
    int bh = blockIdx.z;                      // 0..31
    int b = bh >> 4, h = bh & 15;
    int t = blockIdx.x * 64 + threadIdx.x;    // block (64,4)
    int d = blockIdx.y * 4 + threadIdx.y;
    vT[((size_t)bh * DKH + d) * T_SEQ + t] =
        qkv[(size_t)(b * T_SEQ + t) * (3 * D_MODEL) + 2 * D_MODEL + h * DKH + d];
}

// ---------------------------------------------------------------------------
// Attention: one wave per (16-query tile, head, batch).
// S = Q K^T * 0.125 (WMMA, f16 in LDS), row softmax, O = P V^T (WMMA).
// ---------------------------------------------------------------------------
#define PSTR 2056   // padded LDS row stride in halfs (4112B: 16B aligned, bank-spread)

__global__ __launch_bounds__(32) void attn_wmma(
    const _Float16* __restrict__ qkv, const _Float16* __restrict__ vT,
    _Float16* __restrict__ obuf) {
    __shared__ __align__(16) _Float16 P[16 * PSTR];   // ~64.25 KB of 320 KB/WGP
    int q0 = blockIdx.x * 16;
    int h  = blockIdx.y;
    int b  = blockIdx.z;
    int bh = b * NHEAD + h;
    int lane = threadIdx.x;
    int nl = lane & 15, hv = lane >> 4;

    // Q fragments (row q0+nl, d 0..63) held in registers
    const _Float16* qrow = qkv + (size_t)(b * T_SEQ + q0 + nl) * (3 * D_MODEL) + h * DKH;
    v16h qa0 = frag_a(qrow, hv);
    v16h qa1 = frag_a(qrow + 32, hv);

    // Pass 1: S tiles = Q K^T, scaled, into LDS as f16
    for (int kt = 0; kt < T_SEQ / 16; ++kt) {
        const _Float16* krow =
            qkv + (size_t)(b * T_SEQ + kt * 16 + nl) * (3 * D_MODEL) + D_MODEL + h * DKH;
        v16h kb0 = frag_b(krow, hv);
        v16h kb1 = frag_b(krow + 32, hv);
        v8f s = {};
        s = wmma_f16(qa0, kb0, s);
        s = wmma_f16(qa1, kb1, s);
        for (int j = 0; j < 8; ++j)
            P[(j + 8 * hv) * PSTR + kt * 16 + nl] = (_Float16)(s[j] * 0.125f);
    }
    __syncthreads();

    // Pass 2: row softmax (lane r handles row r); EXEC reconverges after.
    if (lane < 16) {
        _Float16* pr = P + lane * PSTR;
        float mx = -3.4e38f;
        for (int k = 0; k < T_SEQ; ++k) mx = fmaxf(mx, (float)pr[k]);
        float sum = 0.0f;
        for (int k = 0; k < T_SEQ; ++k) {
            float e = __expf((float)pr[k] - mx);
            sum += e;
            pr[k] = (_Float16)e;
        }
        float inv = 1.0f / sum;
        for (int k = 0; k < T_SEQ; ++k) pr[k] = (_Float16)((float)pr[k] * inv);
    }
    __syncthreads();

    // Pass 3: O = P * V  (B fragments from pre-transposed V^T, K-contiguous)
    v8f o0 = {}, o1 = {}, o2 = {}, o3 = {};
    for (int kt = 0; kt < T_SEQ / 32; ++kt) {
        v16h pa = frag_a(&P[nl * PSTR + kt * 32], hv);
        const _Float16* vb = vT + (size_t)bh * DKH * T_SEQ + kt * 32;
        v16h b0 = frag_b(vb + (size_t)(0 * 16 + nl) * T_SEQ, hv);
        v16h b1 = frag_b(vb + (size_t)(1 * 16 + nl) * T_SEQ, hv);
        v16h b2 = frag_b(vb + (size_t)(2 * 16 + nl) * T_SEQ, hv);
        v16h b3 = frag_b(vb + (size_t)(3 * 16 + nl) * T_SEQ, hv);
        o0 = wmma_f16(pa, b0, o0);
        o1 = wmma_f16(pa, b1, o1);
        o2 = wmma_f16(pa, b2, o2);
        o3 = wmma_f16(pa, b3, o3);
    }
    for (int j = 0; j < 8; ++j) {
        size_t row = (size_t)(b * T_SEQ + q0 + j + 8 * hv) * D_MODEL + h * DKH + nl;
        obuf[row + 0 * 16] = (_Float16)o0[j];
        obuf[row + 1 * 16] = (_Float16)o1[j];
        obuf[row + 2 * 16] = (_Float16)o2[j];
        obuf[row + 3 * 16] = (_Float16)o3[j];
    }
}

// ---------------------------------------------------------------------------
// Launcher
// ---------------------------------------------------------------------------
extern "C" void kernel_launch(void* const* d_in, const int* in_sizes, int n_in,
                              void* d_out, int out_size, void* d_ws, size_t ws_size,
                              hipStream_t stream) {
    (void)in_sizes; (void)n_in; (void)out_size; (void)ws_size;
    const float* x     = (const float*)d_in[0];
    const float* w_qkv = (const float*)d_in[1];
    const float* b_qkv = (const float*)d_in[2];
    const float* w_out = (const float*)d_in[3];
    const float* b_out = (const float*)d_in[4];
    const float* w1    = (const float*)d_in[5];
    const float* b1    = (const float*)d_in[6];
    const float* w2    = (const float*)d_in[7];
    const float* b2    = (const float*)d_in[8];
    const float* g1    = (const float*)d_in[9];
    const float* be1   = (const float*)d_in[10];
    const float* g2    = (const float*)d_in[11];
    const float* be2   = (const float*)d_in[12];

    char* w = (char*)d_ws;
    size_t off = 0;
    _Float16* h1    = (_Float16*)(w + off); off += (size_t)MTOT * D_MODEL * 2;        // 8 MB
    _Float16* wqkvT = (_Float16*)(w + off); off += (size_t)3 * D_MODEL * D_MODEL * 2; // 6 MB
    _Float16* qkvb  = (_Float16*)(w + off); off += (size_t)MTOT * 3 * D_MODEL * 2;    // 24 MB
    _Float16* vTb   = (_Float16*)(w + off); off += (size_t)NB * NHEAD * DKH * T_SEQ * 2; // 8 MB
    _Float16* obuf  = (_Float16*)(w + off); off += (size_t)MTOT * D_MODEL * 2;        // 8 MB
    _Float16* woutT = (_Float16*)(w + off); off += (size_t)D_MODEL * D_MODEL * 2;     // 2 MB
    float*    x1    = (float*)   (w + off); off += (size_t)MTOT * D_MODEL * 4;        // 16 MB
    _Float16* h2    = (_Float16*)(w + off); off += (size_t)MTOT * D_MODEL * 2;        // 8 MB
    _Float16* w1T   = (_Float16*)(w + off); off += (size_t)D_FF * D_MODEL * 2;        // 8 MB
    _Float16* hff   = (_Float16*)(w + off); off += (size_t)MTOT * D_FF * 2;           // 32 MB
    _Float16* w2T   = (_Float16*)(w + off); off += (size_t)D_MODEL * D_FF * 2;        // 8 MB

    dim3 t32x8(32, 8);
    // Weight transpose + convert (src[K][N] -> dst[N][K] f16)
    transpose_f32_to_f16<<<dim3(3 * D_MODEL / 32, D_MODEL / 32), t32x8, 0, stream>>>(w_qkv, wqkvT, D_MODEL, 3 * D_MODEL);
    transpose_f32_to_f16<<<dim3(D_MODEL / 32, D_MODEL / 32), t32x8, 0, stream>>>(w_out, woutT, D_MODEL, D_MODEL);
    transpose_f32_to_f16<<<dim3(D_FF / 32, D_MODEL / 32), t32x8, 0, stream>>>(w1, w1T, D_MODEL, D_FF);
    transpose_f32_to_f16<<<dim3(D_MODEL / 32, D_FF / 32), t32x8, 0, stream>>>(w2, w2T, D_FF, D_MODEL);

    // ln1 -> h1 (f16)
    layernorm_f16<<<MTOT, 256, 0, stream>>>(x, g1, be1, h1);

    // qkv = h1 @ w_qkv + b_qkv  -> f16 [4096][3072]
    gemm_wmma<0><<<dim3(3 * D_MODEL / 64, MTOT / 64), 256, 0, stream>>>(
        h1, wqkvT, b_qkv, nullptr, qkvb, MTOT, 3 * D_MODEL, D_MODEL);

    // V^T extraction for the PV WMMA
    extract_vT<<<dim3(T_SEQ / 64, NHEAD, NB * NHEAD), dim3(64, 4), 0, stream>>>(qkvb, vTb);

    // attention -> obuf (f16 [4096][1024], heads concatenated)
    attn_wmma<<<dim3(T_SEQ / 16, NHEAD, NB), 32, 0, stream>>>(qkvb, vTb, obuf);

    // x1 = x + obuf @ w_out + b_out  (f32)
    gemm_wmma<2><<<dim3(D_MODEL / 64, MTOT / 64), 256, 0, stream>>>(
        obuf, woutT, b_out, x, x1, MTOT, D_MODEL, D_MODEL);

    // ln2 -> h2 (f16)
    layernorm_f16<<<MTOT, 256, 0, stream>>>(x1, g2, be2, h2);

    // hff = gelu(h2 @ w1 + b1) -> f16 [4096][4096]
    gemm_wmma<1><<<dim3(D_FF / 64, MTOT / 64), 256, 0, stream>>>(
        h2, w1T, b1, nullptr, hff, MTOT, D_FF, D_MODEL);

    // out = x1 + hff @ w2 + b2  (f32)
    gemm_wmma<2><<<dim3(D_MODEL / 64, MTOT / 64), 256, 0, stream>>>(
        hff, w2T, b2, x1, (float*)d_out, MTOT, D_MODEL, D_FF);
}
